// MultiHeadAttentionModule_30923764531694
// MI455X (gfx1250) — matrix operand order repaired and checked by reading, not compiled, not run on hardware
//
#include <hip/hip_runtime.h>

// ---------------------------------------------------------------------------
// MI455X / gfx1250 cross-attention:  (GLO', LOC') = CrossAttn(GLO,LOC) both ways
// D=1024, H=16, HD=64, S=2048, B=1.  bf16 WMMA (f32 accum) throughout.
// Data movement: TDM tensor_load_to_lds for strided 2-D tiles (A / K tiles),
// transposed LDS layouts so all WMMA fragment loads are 2x ds_load_b128.
// ---------------------------------------------------------------------------

#define DD   1024
#define HH   16
#define HDD  64
#define SS   2048

typedef __attribute__((ext_vector_type(16))) __bf16          v16bf;
typedef __attribute__((ext_vector_type(16))) unsigned short  v16us;
typedef __attribute__((ext_vector_type(8)))  float           v8f;
typedef __attribute__((ext_vector_type(4)))  unsigned int    v4u;
typedef __attribute__((ext_vector_type(8)))  int             v8i;
typedef __attribute__((ext_vector_type(4)))  int             v4i;

__device__ __forceinline__ unsigned short f32_to_bf16_rne(float f) {
    unsigned int u = __builtin_bit_cast(unsigned int, f);
    unsigned int r = u + 0x7FFFu + ((u >> 16) & 1u);
    return (unsigned short)(r >> 16);
}

// ---------------------------------------------------------------------------
// Tensor Data Mover: 2-D tile (elem=2B) global -> LDS, row-major into LDS.
// D# built per ISA 8.3/8.4:  group0 = {count, lds_addr, global_addr, type=2},
// group1 = {data_size, tensor_dim0/1, tile_dim0/1, tensor_dim0_stride}.
// Issue from ONE wave; completion via TENSORcnt.
// ---------------------------------------------------------------------------
__device__ __forceinline__ void tdm_load_2d_bf16(unsigned lds_addr, const void* gptr,
                                                 unsigned dim0, unsigned dim1,
                                                 unsigned stride0) {
    unsigned long long ga = (unsigned long long)gptr;
    v4u g0;
    g0[0] = 1u;                                            // count=1, user-mode
    g0[1] = lds_addr;                                      // LDS byte address
    g0[2] = (unsigned)(ga & 0xFFFFFFFFu);                  // global_addr[31:0]
    g0[3] = (unsigned)((ga >> 32) & 0x01FFFFFFu) | (2u << 30); // addr[56:32] | type=2
    v8i g1;
    g1[0] = (int)(1u << 16);                               // data_size=1 (2 bytes)
    g1[1] = (int)((dim0 & 0xFFFFu) << 16);                 // tensor_dim0[15:0]
    g1[2] = (int)(((dim0 >> 16) & 0xFFFFu) | ((dim1 & 0xFFFFu) << 16)); // dim0 hi | dim1 lo
    g1[3] = (int)(((dim1 >> 16) & 0xFFFFu) | ((dim0 & 0xFFFFu) << 16)); // dim1 hi | tile_dim0
    g1[4] = (int)(dim1 & 0xFFFFu);                         // tile_dim1 (tile==tensor)
    g1[5] = (int)stride0;                                  // tensor_dim0_stride lo32
    g1[6] = 0;
    g1[7] = 0;
    v4i z4 = {0, 0, 0, 0};                                 // groups 2/3 unused (2-D)
#if defined(__clang_major__) && (__clang_major__ >= 23)
    v8i z8 = {0, 0, 0, 0, 0, 0, 0, 0};
    __builtin_amdgcn_tensor_load_to_lds(g0, g1, z4, z4, z8, 0);
#else
    __builtin_amdgcn_tensor_load_to_lds(g0, g1, z4, z4, 0);
#endif
}

// ---- WMMA fragment loaders (contiguous k order, same for A and B, wave32) --------------------
// A (16x32 MxK, bf16): lanes 0-15 -> M=lane,  K = 0..15
//                      lanes16-31 -> M=lane-16, K = 16..31
// Source row-major [rows][ld]; per-lane reads are two 16-byte runs.
__device__ __forceinline__ v16bf load_A_frag(const unsigned short* base,
                                             int row0, int ld, int k0) {
    const int lane = threadIdx.x & 31;
    const int m    = lane & 15;
    const int kh   = (lane >> 4) << 4;           // 0 or 16
    const unsigned short* p = base + (long)(row0 + m) * ld + k0;
    v16us t;
#pragma unroll
    for (int i = 0; i < 8; ++i) t[i]     = p[kh + i];
#pragma unroll
    for (int i = 0; i < 8; ++i) t[8 + i] = p[kh + 8 + i];
    return __builtin_bit_cast(v16bf, t);
}

// B (32x16 KxN) from an LDS tile stored K-contiguous per column: base[n][k],
// row pitch ldk.  lanes 0-15 -> N=lane, K=0..15 ; lanes 16-31 -> N, K=16..31.
// 32 contiguous bytes per lane -> 2x ds_load_b128.
__device__ __forceinline__ v16bf load_B_fragT(const unsigned short* base,
                                              int n0, int ldk, int k0) {
    const int lane = threadIdx.x & 31;
    const int n    = lane & 15;
    const int kh   = (lane >> 4) << 4;           // 0 or 16
    const unsigned short* p = base + (long)(n0 + n) * ldk + k0 + kh;
    v16us t;
#pragma unroll
    for (int i = 0; i < 16; ++i) t[i] = p[i];
    return __builtin_bit_cast(v16bf, t);
}

__device__ __forceinline__ void store_val(float* p, float v)          { *p = v; }
__device__ __forceinline__ void store_val(unsigned short* p, float v) { *p = f32_to_bf16_rne(v); }

// ---------------------------------------------------------------------------
// f32 -> bf16 conversion
// ---------------------------------------------------------------------------
__global__ __launch_bounds__(256)
void cvt_f32_to_bf16_kernel(const float* __restrict__ in,
                            unsigned short* __restrict__ out, int n) {
    int i = blockIdx.x * 256 + threadIdx.x;
    if (i < n) out[i] = f32_to_bf16_rne(in[i]);
}

// ---------------------------------------------------------------------------
// Generic batched bf16 GEMM:  C[b] = A @ B[b]      (M x K) x (K x N)
// block = 256 threads = 8 waves; tile BM=128, BN=64, BK=32.
// A tile: TDM load (row-major [128][32]).  B tile: transposed LDS [64][32].
// wave w computes rows [w*16, w*16+16) x full BN via 4 accumulators.
// ---------------------------------------------------------------------------
template <typename OutT>
__global__ __launch_bounds__(256)
void gemm_bf16_kernel(const unsigned short* __restrict__ A, int lda,
                      const unsigned short* __restrict__ B, int ldb, long strideB,
                      OutT* __restrict__ C, int ldc, long strideC,
                      int M, int N, int K) {
    constexpr int BM = 128, BN = 64, BK = 32;
    __shared__ unsigned short Asm[BM * BK];   // 8 KB, row-major [BM][BK]
    __shared__ unsigned short Bt [BN * BK];   // 4 KB, transposed  [BN][BK]

    const int ntiles = N / BN;
    const int mt = blockIdx.x / ntiles;
    const int nt = blockIdx.x % ntiles;
    const int m0 = mt * BM;
    const int n0 = nt * BN;

    const unsigned short* Bb = B + (long)blockIdx.y * strideB;
    OutT*                 Cb = C + (long)blockIdx.y * strideC;

    const int tid  = threadIdx.x;
    const int wave = tid >> 5;
    const int lane = tid & 31;

    const unsigned asm_lds = (unsigned)(unsigned long long)(&Asm[0]);

    v8f acc[4] = {};

    for (int k0 = 0; k0 < K; k0 += BK) {
        // -- A tile via Tensor Data Mover: tile0=BK elems, tile1=BM rows --
        if (wave == 0)
            tdm_load_2d_bf16(asm_lds, &A[(long)m0 * lda + k0], BK, BM, (unsigned)lda);

        // -- B tile, transposed store: Bt[n][k] = B[k0+k][n0+n] --
#pragma unroll
        for (int i = 0; i < 4; ++i) {           // 1024 (n,kpair) units
            int p  = tid + i * 256;
            int n  = p & 63;
            int k  = (p >> 6) << 1;
            unsigned lo = Bb[(long)(k0 + k)     * ldb + n0 + n];
            unsigned hi = Bb[(long)(k0 + k + 1) * ldb + n0 + n];
            *(unsigned int*)&Bt[n * BK + k] = lo | (hi << 16);
        }
        if (k0 + BK < K)                        // gfx1250 global_prefetch_b8
            __builtin_prefetch(&Bb[(long)(k0 + BK + (tid >> 3)) * ldb + n0], 0, 1);

        if (wave == 0) __builtin_amdgcn_s_wait_tensorcnt(0);
        __syncthreads();

        v16bf a = load_A_frag(Asm, wave * 16, BK, 0);
#pragma unroll
        for (int nn = 0; nn < 4; ++nn) {
            v16bf b = load_B_fragT(Bt, nn * 16, BK, 0);
            acc[nn] = __builtin_amdgcn_wmma_f32_16x16x32_bf16(
                false, a, false, b, (short)0, acc[nn], false, false);
        }
        __syncthreads();
    }

    // -- epilogue: C layout  VGPR j: lanes0-15 -> (M=j,N=lane), 16-31 -> (M=j+8,N=lane-16)
    const int rbase = m0 + wave * 16 + ((lane >> 4) << 3);
    const int cbase = n0 + (lane & 15);
#pragma unroll
    for (int nn = 0; nn < 4; ++nn)
#pragma unroll
        for (int j = 0; j < 8; ++j)
            store_val(&Cb[(long)(rbase + j) * ldc + cbase + nn * 16], acc[nn][j]);
}

// ---------------------------------------------------------------------------
// Flash attention.  grid = (S/128, H, 2 dirs), block = 256 (8 waves).
// proj layout: [src][kind][head][S][HD]  (kind 0=Q,1=K,2=V)
// attn_out layout: [dir][H][S][HD] flat  (== torch .view reinterpretation)
// K tile: TDM row-major [32][64] (score B-frags read hd-contiguous).
// V tile: transposed LDS [64 hd][32 key] (PV B-frags read key-contiguous).
// ---------------------------------------------------------------------------
__global__ __launch_bounds__(256)
void flash_attn_kernel(const unsigned short* __restrict__ proj,
                       unsigned short* __restrict__ attn_out) {
    __shared__ unsigned short Ksm[32 * HDD];        // 4 KB [key][hd]
    __shared__ unsigned short Vt [HDD * 32];        // 4 KB [hd][key]
    __shared__ unsigned short Psm[8][16 * 32];      // 8 KB per-wave P repack

    const int dir  = blockIdx.z;
    const int h    = blockIdx.y;
    const int tid  = threadIdx.x;
    const int wave = tid >> 5;
    const int lane = tid & 31;
    const int qrow = blockIdx.x * 128 + wave * 16;

    const long planeSz = (long)SS * HDD;
    const unsigned short* Qg = proj + (((long)dir * 3 + 0) * HH + h) * planeSz;
    const unsigned short* Kg = proj + (((long)(1 - dir) * 3 + 1) * HH + h) * planeSz;
    const unsigned short* Vg = proj + (((long)(1 - dir) * 3 + 2) * HH + h) * planeSz;

    const unsigned ksm_lds = (unsigned)(unsigned long long)(&Ksm[0]);

    // Q tile 16x64 -> two A fragments, kept in registers for all 64 key blocks
    v16bf qa0 = load_A_frag(Qg, qrow, HDD, 0);
    v16bf qa1 = load_A_frag(Qg, qrow, HDD, 32);

    float m[8], l[8];
#pragma unroll
    for (int j = 0; j < 8; ++j) { m[j] = -1e30f; l[j] = 0.0f; }
    v8f O[4] = {};

    for (int kb = 0; kb < SS; kb += 32) {
        // -- K block via TDM: 32 rows x 64 elems, row pitch 64 --
        if (wave == 0)
            tdm_load_2d_bf16(ksm_lds, &Kg[(long)kb * HDD], HDD, 32, HDD);

        // -- V block, transposed store: Vt[e][k] = V[kb+k][e] --
#pragma unroll
        for (int i = 0; i < 4; ++i) {           // 1024 (e,kpair) units
            int p = tid + i * 256;
            int e = p & 63;
            int k = (p >> 6) << 1;
            unsigned lo = Vg[(long)(kb + k)     * HDD + e];
            unsigned hi = Vg[(long)(kb + k + 1) * HDD + e];
            *(unsigned int*)&Vt[e * 32 + k] = lo | (hi << 16);
        }
        if (kb + 32 < SS) {
            __builtin_prefetch(&Kg[(long)(kb + 32 + (tid >> 3)) * HDD], 0, 1);
            __builtin_prefetch(&Vg[(long)(kb + 32 + (tid >> 3)) * HDD], 0, 1);
        }
        if (wave == 0) __builtin_amdgcn_s_wait_tensorcnt(0);
        __syncthreads();

        // -- scores: Q(16x64) @ K^T(64x32): B-frag (k=hd, n=key) is
        //    hd-contiguous per lane in the row-major K tile --
        v8f st[2];
#pragma unroll
        for (int t = 0; t < 2; ++t) {
            v8f s = {};
            s = __builtin_amdgcn_wmma_f32_16x16x32_bf16(
                false, qa0, false, load_B_fragT(Ksm, t * 16, HDD, 0),
                (short)0, s, false, false);
            s = __builtin_amdgcn_wmma_f32_16x16x32_bf16(
                false, qa1, false, load_B_fragT(Ksm, t * 16, HDD, 32),
                (short)0, s, false, false);
#pragma unroll
            for (int j = 0; j < 8; ++j) s[j] *= 0.125f;   // 1/sqrt(64)
            st[t] = s;
        }

        // -- online softmax; each row lives fully in one 16-lane half --
        unsigned short* Pw = &Psm[wave][0];
        float sc[8];
#pragma unroll
        for (int j = 0; j < 8; ++j) {
            float v = fmaxf(st[0][j], st[1][j]);
#pragma unroll
            for (int off = 8; off >= 1; off >>= 1) v = fmaxf(v, __shfl_xor(v, off));
            float mn = fmaxf(m[j], v);
            sc[j] = __expf(m[j] - mn);
            m[j]  = mn;
        }
#pragma unroll
        for (int j = 0; j < 8; ++j) {
            float p0 = __expf(st[0][j] - m[j]);
            float p1 = __expf(st[1][j] - m[j]);
            float rs = p0 + p1;
#pragma unroll
            for (int off = 8; off >= 1; off >>= 1) rs += __shfl_xor(rs, off);
            l[j] = l[j] * sc[j] + rs;
            const int row  = j + ((lane >> 4) << 3);
            const int key0 = lane & 15;
            Pw[row * 32 + key0]      = f32_to_bf16_rne(p0);
            Pw[row * 32 + 16 + key0] = f32_to_bf16_rne(p1);
#pragma unroll
            for (int nn = 0; nn < 4; ++nn) O[nn][j] *= sc[j];
        }
        __syncthreads();   // make P visible in A-fragment order

        // -- O += P(16x32) @ V(32x64): B-frag (k=key, n=hd) is key-contiguous
        //    per lane in the transposed V tile --
        v16bf pa = load_A_frag(Pw, 0, 32, 0);
#pragma unroll
        for (int nn = 0; nn < 4; ++nn) {
            v16bf vb = load_B_fragT(Vt, nn * 16, 32, 0);
            O[nn] = __builtin_amdgcn_wmma_f32_16x16x32_bf16(
                false, pa, false, vb, (short)0, O[nn], false, false);
        }
        __syncthreads();   // before K/V overwrite next iteration
    }

    // -- normalize and store:  attn_out[dir][h][q][e] --
    float invl[8];
#pragma unroll
    for (int j = 0; j < 8; ++j) invl[j] = 1.0f / l[j];
    unsigned short* Ob = attn_out + (((long)dir * HH + h) * SS) * HDD;
    const int rbase = qrow + ((lane >> 4) << 3);
#pragma unroll
    for (int nn = 0; nn < 4; ++nn)
#pragma unroll
        for (int j = 0; j < 8; ++j)
            Ob[(long)(rbase + j) * HDD + nn * 16 + (lane & 15)] =
                f32_to_bf16_rne(O[nn][j] * invl[j]);
}

// ---------------------------------------------------------------------------
// Launcher
// ---------------------------------------------------------------------------
extern "C" void kernel_launch(void* const* d_in, const int* in_sizes, int n_in,
                              void* d_out, int out_size, void* d_ws, size_t ws_size,
                              hipStream_t stream) {
    (void)in_sizes; (void)n_in; (void)out_size; (void)ws_size;

    const float* GLO = (const float*)d_in[0];
    const float* LOC = (const float*)d_in[1];
    const float* W_q = (const float*)d_in[2];
    const float* W_k = (const float*)d_in[3];
    const float* W_v = (const float*)d_in[4];
    const float* W_o = (const float*)d_in[5];
    float* out = (float*)d_out;

    // workspace layout (ushort elements)
    unsigned short* ws = (unsigned short*)d_ws;
    const long nX  = (long)SS * DD;            // per source
    const long nW  = (long)HH * DD * HDD;      // per qkv weight
    const long nWo = (long)DD * DD;
    const long nPl = (long)SS * HDD;           // per (src,kind,head) plane

    unsigned short* Xbf   = ws;                        // [2][S*D]
    unsigned short* Wqb   = Xbf  + 2 * nX;
    unsigned short* Wkb   = Wqb  + nW;
    unsigned short* Wvb   = Wkb  + nW;
    unsigned short* Wob   = Wvb  + nW;
    unsigned short* projB = Wob  + nWo;                // [2][3][H][S][HD]
    unsigned short* AO    = projB + 6 * HH * nPl;      // [2][S*D]

    // 1) f32 -> bf16 conversions
    auto cvt = [&](const float* src, unsigned short* dst, long n) {
        cvt_f32_to_bf16_kernel<<<dim3((unsigned)((n + 255) / 256)), 256, 0, stream>>>(
            src, dst, (int)n);
    };
    cvt(GLO, Xbf,      nX);
    cvt(LOC, Xbf + nX, nX);
    cvt(W_q, Wqb, nW);
    cvt(W_k, Wkb, nW);
    cvt(W_v, Wvb, nW);
    cvt(W_o, Wob, nWo);

    // 2) Q/K/V projections: per (src,kind), batched over heads
    const unsigned short* Wtab[3] = { Wqb, Wkb, Wvb };
    for (int src = 0; src < 2; ++src) {
        for (int kind = 0; kind < 3; ++kind) {
            const unsigned short* A = Xbf + (long)src * nX;
            const unsigned short* B = Wtab[kind];
            unsigned short* C = projB + (long)(src * 3 + kind) * HH * nPl;
            dim3 grid((SS / 128) * (HDD / 64), HH);
            gemm_bf16_kernel<unsigned short><<<grid, 256, 0, stream>>>(
                A, DD, B, HDD, (long)DD * HDD, C, HDD, nPl, SS, HDD, DD);
        }
    }

    // 3) flash attention, both directions
    {
        dim3 grid(SS / 128, HH, 2);
        flash_attn_kernel<<<grid, 256, 0, stream>>>(projB, AO);
    }

    // 4) output projection: out[dir] = AO[dir] (2048x1024) @ W_o (1024x1024)
    for (int dir = 0; dir < 2; ++dir) {
        const unsigned short* A = AO + (long)dir * nX;
        float* C = out + (long)dir * nX;
        dim3 grid((SS / 128) * (DD / 64), 1);
        gemm_bf16_kernel<float><<<grid, 256, 0, stream>>>(
            A, DD, Wob, DD, 0L, C, DD, 0L, SS, DD, DD);
    }
}
